// SpatialBlock_68513318306528
// MI455X (gfx1250) — matrix operand order repaired and checked by reading, not compiled, not run on hardware
//
#include <hip/hip_runtime.h>
#include <hip/hip_bf16.h>
#include <math.h>

// Problem constants (match reference)
#define BB   2
#define TT   2048
#define DD   1024
#define HH   16
#define DHD  64
#define D3   3072
#define DF   4096
#define NT_T (TT/16)        // 128 token tiles
#define MT_ALL ((BB*TT)/16) // 256 row tiles over (B*T)
#define TRI  (NT_T*(NT_T+1)/2) // 8256 lower-triangle tiles per (b,h)
#define STG_K 64            // K-depth per async stage

typedef __bf16 bf16;
typedef __attribute__((ext_vector_type(16))) __bf16 v16bf;
typedef __attribute__((ext_vector_type(8)))  __bf16 v8bf;
typedef __attribute__((ext_vector_type(8)))  float  v8f;
typedef __attribute__((ext_vector_type(4)))  unsigned int u32x4;
typedef __attribute__((ext_vector_type(8)))  unsigned int u32x8;

// ---------------- WMMA fragment loaders (CDNA5 16-bit layouts) ----------------
// A 16x32 bf16: lane&15 = M row; lane>>4 selects K half. Chunks at K = h*8 and 16+h*8.
__device__ __forceinline__ v16bf ld_a_bf16(const bf16* base, int lda, int lane) {
    int m  = lane & 15;
    int hh = lane >> 4;
    const bf16* p = base + (size_t)m * lda + hh * 8;
    v8bf lo = *(const v8bf*)p;
    v8bf hi = *(const v8bf*)(p + 16);
    v16bf r;
#pragma unroll
    for (int i = 0; i < 8; ++i) { r[i] = lo[i]; r[8 + i] = hi[i]; }
    return r;
}

// A 16x32 from fp32 source (attention weights), converted to bf16 on the fly.
__device__ __forceinline__ v16bf ld_a_f32(const float* base, int lda, int lane) {
    int m  = lane & 15;
    int hh = lane >> 4;
    const float* p = base + (size_t)m * lda + hh * 8;
    v16bf r;
#pragma unroll
    for (int i = 0; i < 8; ++i) { r[i] = (bf16)p[i]; r[8 + i] = (bf16)p[16 + i]; }
    return r;
}

// B 32x16 bf16 loaded from B^T row-major storage: lane&15 = N column (a row of B^T),
// lane>>4 selects K half. Chunks at K = h*8 and 16+h*8.
__device__ __forceinline__ v16bf ld_b_bf16(const bf16* baseT, int ldb, int lane) {
    int n  = lane & 15;
    int hh = lane >> 4;
    const bf16* p = baseT + (size_t)n * ldb + hh * 8;
    v8bf lo = *(const v8bf*)p;
    v8bf hi = *(const v8bf*)(p + 16);
    v16bf r;
#pragma unroll
    for (int i = 0; i < 8; ++i) { r[i] = lo[i]; r[8 + i] = hi[i]; }
    return r;
}

// Direct-from-global K-loop (used by the attention kernels).
__device__ __forceinline__ v8f wmma_loop_bf16(const bf16* A, int lda,
                                              const bf16* Bt, int ldb,
                                              int kTiles, int lane, v8f acc) {
    for (int kk = 0; kk < kTiles; ++kk) {
        const bf16* ap = A  + kk * 32;
        const bf16* bp = Bt + kk * 32;
        __builtin_prefetch(ap + 64, 0, 3);
        __builtin_prefetch(bp + 64, 0, 3);
        v16bf a = ld_a_bf16(ap, lda, lane);
        v16bf b = ld_b_bf16(bp, ldb, lane);
        acc = __builtin_amdgcn_wmma_f32_16x16x32_bf16(false, a, false, b,
                                                      (short)0, acc, false, false);
    }
    return acc;
}

// C layout: VGPR r, lanes 0-15 -> M=r, N=lane; lanes 16-31 -> M=8+r, N=lane-16.
#define C_MLOC(lane) (((lane) >> 4) * 8)
#define C_NLOC(lane) ((lane) & 15)

// ---------------- CDNA5 async Global->LDS + TDM staging ----------------
__device__ __forceinline__ void async_ld_b128(const bf16* gptr, unsigned ldsoff) {
    // ASYNCcnt-tracked DMA of 16B/lane into LDS (CDNA5 ISA)
    asm volatile("global_load_async_to_lds_b128 %0, %1, off"
                 :: "v"(ldsoff), "v"(gptr)
                 : "memory");
}
__device__ __forceinline__ void wait_async_le2() {
    asm volatile("s_wait_asynccnt 0x2" ::: "memory");
}
__device__ __forceinline__ void wait_async_0() {
    asm volatile("s_wait_asynccnt 0x0" ::: "memory");
}
#define LDSOFF(p) ((unsigned)(uintptr_t)(p))

// TDM: load a 64(rows) x 64(K elems) bf16 tile (row stride = ldk elems) into LDS.
// 2D D# -> two SGPR groups (VADDR2/3 NULL). TENSORcnt-tracked. Wave-level op.
__device__ __forceinline__ void tdm_load_tile64x64(const bf16* gptr, unsigned ldsoff,
                                                   unsigned ldk) {
    unsigned long long ga = (unsigned long long)(uintptr_t)gptr;
    // group0: [1:0]=count=1 | lds_addr[63:32] | global_addr[120:64] | type=2 [127:126]
    u32x4 g0 = { 1u,
                 ldsoff,
                 (unsigned)ga,
                 (unsigned)((ga >> 32) & 0x01FFFFFFu) | (2u << 30) };
    // group1: data_size=2B; tensor_dim0=64, tensor_dim1=64; tile 64x64; stride0=ldk
    u32x8 g1 = { 0x00010000u,        // data_size=1 (2 bytes) at bits[17:16]
                 64u << 16,          // tensor_dim0[15:0] -> bits[79:64+? g1[31:16]]
                 64u << 16,          // tensor_dim0[31:16]=0 | tensor_dim1[15:0]<<16
                 64u << 16,          // tensor_dim1[31:16]=0 | tile_dim0=64 <<16
                 64u,                // tile_dim1=64 | tile_dim2=0
                 ldk,                // tensor_dim0_stride[31:0]
                 0u, 0u };
    asm volatile("tensor_load_to_lds %0, %1" :: "s"(g0), "s"(g1) : "memory");
}

// LDS fragment readers (row stride = STG_K) -> ds_load_b128
__device__ __forceinline__ v16bf frag_a_lds(const bf16* p, int lane) {
    int m = lane & 15, hh = lane >> 4;
    const bf16* q = p + m * STG_K + hh * 8;
    v8bf lo = *(const v8bf*)q;
    v8bf hi = *(const v8bf*)(q + 16);
    v16bf r;
#pragma unroll
    for (int i = 0; i < 8; ++i) { r[i] = lo[i]; r[8 + i] = hi[i]; }
    return r;
}
__device__ __forceinline__ v16bf frag_b_lds(const bf16* p, int lane) {
    int n = lane & 15, hh = lane >> 4;
    const bf16* q = p + n * STG_K + hh * 8;
    v8bf lo = *(const v8bf*)q;
    v8bf hi = *(const v8bf*)(q + 16);
    v16bf r;
#pragma unroll
    for (int i = 0; i < 8; ++i) { r[i] = lo[i]; r[8 + i] = hi[i]; }
    return r;
}

// ---------------- staged macro-tile GEMM: 64M x 64N per block ----------------
// A staged by per-lane async-to-LDS; B staged by TDM (wave 0). Double-buffered.
// EPI: 0 = proj (out = xin + acc), 1 = ffn1 (bf16 gelu(acc+bias)),
//      2 = ffn2 (outF += acc + bias), 3 = qkv scatter to q/k/v + vT
template <int EPI>
__global__ __launch_bounds__(256) void k_gemm_staged(
    const bf16* __restrict__ A, const bf16* __restrict__ Bt, int K, int Nall,
    const float* __restrict__ xin, const float* __restrict__ bias,
    float* __restrict__ outF, bf16* __restrict__ outB,
    bf16* __restrict__ qh, bf16* __restrict__ kh,
    bf16* __restrict__ vh, bf16* __restrict__ vT) {
    __shared__ __attribute__((aligned(16))) bf16 As[2][64][STG_K];
    __shared__ __attribute__((aligned(16))) bf16 Bs[2][64][STG_K];
    int tid  = threadIdx.x;
    int lane = tid & 31;
    int w    = tid >> 5;                 // 8 waves
    int mblk = blockIdx.x;               // M/64 blocks
    int nblk = blockIdx.y;               // N/64 blocks
    const bf16* Ag = A  + (size_t)mblk * 64 * K;
    const bf16* Bg = Bt + (size_t)nblk * 64 * K;
    int nstages = K / STG_K;

    // each thread stages exactly 2 b128 chunks of A per stage
    auto stage_issue_A = [&](int k0, int buf) {
#pragma unroll
        for (int u = 0; u < 2; ++u) {
            int c   = tid + u * 256;     // 512 chunks
            int row = c >> 3;
            int off = (c & 7) * 8;
            async_ld_b128(Ag + (size_t)row * K + k0 + off, LDSOFF(&As[buf][row][off]));
        }
    };

    v8f acc0 = {}, acc1 = {};
    int mi0 = (w >> 2) * 2;              // two M tiles per wave
    int nw  = w & 3;                     // one N tile per wave

    stage_issue_A(0, 0);
    if (w == 0) tdm_load_tile64x64(Bg, LDSOFF(&Bs[0][0][0]), (unsigned)K);
    for (int s = 0; s < nstages; ++s) {
        int buf = s & 1;
        if (s + 1 < nstages) {
            stage_issue_A((s + 1) * STG_K, buf ^ 1);
            if (w == 0)
                tdm_load_tile64x64(Bg + (s + 1) * STG_K,
                                   LDSOFF(&Bs[buf ^ 1][0][0]), (unsigned)K);
            wait_async_le2();            // previous A stage landed; next in flight
            if (w == 0) __builtin_amdgcn_s_wait_tensorcnt((short)1);
        } else {
            wait_async_0();
            if (w == 0) __builtin_amdgcn_s_wait_tensorcnt((short)0);
        }
        __syncthreads();
#pragma unroll
        for (int ks = 0; ks < STG_K; ks += 32) {
            v16bf b  = frag_b_lds(&Bs[buf][nw * 16][ks], lane);
            v16bf a0 = frag_a_lds(&As[buf][mi0 * 16][ks], lane);
            v16bf a1 = frag_a_lds(&As[buf][(mi0 + 1) * 16][ks], lane);
            acc0 = __builtin_amdgcn_wmma_f32_16x16x32_bf16(false, a0, false, b,
                                                           (short)0, acc0, false, false);
            acc1 = __builtin_amdgcn_wmma_f32_16x16x32_bf16(false, a1, false, b,
                                                           (short)0, acc1, false, false);
        }
        __syncthreads();                 // all waves done reading buf before overwrite
    }

    int mloc = C_MLOC(lane), nloc = C_NLOC(lane);
    int n0 = nblk * 64 + nw * 16 + nloc;
#pragma unroll
    for (int ti = 0; ti < 2; ++ti) {
        v8f acc = ti ? acc1 : acc0;
        int mbase = mblk * 64 + (mi0 + ti) * 16 + mloc;
        if (EPI == 0) {
#pragma unroll
            for (int i = 0; i < 8; ++i) {
                size_t idx = (size_t)(mbase + i) * Nall + n0;
                outF[idx] = xin[idx] + acc[i];
            }
        } else if (EPI == 1) {
            float bv = bias[n0];
#pragma unroll
            for (int i = 0; i < 8; ++i) {
                float v = acc[i] + bv;
                float g = 0.5f * v * (1.0f + erff(v * 0.70710678f));
                outB[(size_t)(mbase + i) * Nall + n0] = (bf16)g;
            }
        } else if (EPI == 2) {
            float bv = bias[n0];
#pragma unroll
            for (int i = 0; i < 8; ++i) {
                size_t idx = (size_t)(mbase + i) * Nall + n0;
                outF[idx] = outF[idx] + acc[i] + bv;
            }
        } else { // EPI == 3: QKV scatter
            int which = n0 / DD;         // 0=q 1=k 2=v (uniform per tile)
            int r = n0 % DD;
            int head = r / DHD, dh = r % DHD;
#pragma unroll
            for (int i = 0; i < 8; ++i) {
                int mg = mbase + i;
                int b = mg / TT, t = mg % TT;
                size_t idx = (((size_t)b * HH + head) * TT + t) * DHD + dh;
                bf16 val = (bf16)acc[i];
                if (which == 0)      qh[idx] = val;
                else if (which == 1) kh[idx] = val;
                else {
                    vh[idx] = val;
                    vT[(((size_t)b * HH + head) * DHD + dh) * TT + t] = val;
                }
            }
        }
    }
}

// ---------------- weight prep: fp32 [K][N] -> bf16 [N][K] ----------------
__global__ void k_prep_wT(const float* __restrict__ src, bf16* __restrict__ dst,
                          int K, int N) {
    size_t idx = (size_t)blockIdx.x * blockDim.x + threadIdx.x;
    size_t tot = (size_t)K * N;
    if (idx >= tot) return;
    int n = (int)(idx % N);
    int k = (int)(idx / N);
    dst[(size_t)n * K + k] = (bf16)src[idx];
}

// ---------------- layernorm (fp32 in -> bf16 out), one block per token ----------------
__global__ void k_layernorm(const float* __restrict__ x, const float* __restrict__ g,
                            const float* __restrict__ bta, bf16* __restrict__ out) {
    int t = blockIdx.x;                 // 0 .. B*T-1
    const float* row = x + (size_t)t * DD;
    __shared__ float red[256];
    int tid = threadIdx.x;
    float s = 0.f;
    for (int i = tid; i < DD; i += 256) s += row[i];
    red[tid] = s; __syncthreads();
    for (int off = 128; off > 0; off >>= 1) {
        if (tid < off) red[tid] += red[tid + off];
        __syncthreads();
    }
    float mu = red[0] * (1.0f / DD);
    __syncthreads();
    float v = 0.f;
    for (int i = tid; i < DD; i += 256) { float d = row[i] - mu; v += d * d; }
    red[tid] = v; __syncthreads();
    for (int off = 128; off > 0; off >>= 1) {
        if (tid < off) red[tid] += red[tid + off];
        __syncthreads();
    }
    float inv = rsqrtf(red[0] * (1.0f / DD) + 1e-5f);
    __syncthreads();
    for (int i = tid; i < DD; i += 256)
        out[(size_t)t * DD + i] = (bf16)((row[i] - mu) * inv * g[i] + bta[i]);
}

// ---------------- logits: per (b,h) lower-triangle 16x16 tiles of Q K^T ----------------
__global__ void k_logits(const bf16* __restrict__ qh, const bf16* __restrict__ kh,
                         const float* __restrict__ dists, float* __restrict__ wbuf) {
    int wave = blockIdx.x * (blockDim.x >> 5) + (threadIdx.x >> 5);
    int lane = threadIdx.x & 31;
    int bh = wave / TRI;
    if (bh >= BB * HH) return;
    int tri = wave % TRI;
    int it = (int)((sqrtf(8.0f * (float)tri + 1.0f) - 1.0f) * 0.5f);
    while ((it + 1) * (it + 2) / 2 <= tri) ++it;
    while (it * (it + 1) / 2 > tri) --it;
    int jt = tri - it * (it + 1) / 2;    // jt <= it
    int b = bh / HH;
    const bf16* Q = qh + ((size_t)bh * TT + it * 16) * DHD;
    const bf16* K = kh + ((size_t)bh * TT + jt * 16) * DHD;   // rows of K == cols of K^T
    v8f acc = {};
    acc = wmma_loop_bf16(Q, DHD, K, DHD, DHD / 32, lane, acc);
    int mloc = C_MLOC(lane), nloc = C_NLOC(lane);
    int jg = jt * 16 + nloc;
#pragma unroll
    for (int i = 0; i < 8; ++i) {
        int ig = it * 16 + mloc + i;
        float v = acc[i] * 0.125f        // 1/sqrt(64)
                - dists[((size_t)b * TT + ig) * TT + jg];   // PENALTY = 1.0
        wbuf[((size_t)bh * TT + ig) * TT + jg] = v;
    }
}

// ---------------- softmax + energies, one 256-thread block per row ----------------
__global__ void k_softmax(float* __restrict__ wbuf, const float* __restrict__ dists,
                          float* __restrict__ scal) {
    int row = blockIdx.x;                // 0 .. B*H*T-1
    int i  = row % TT;
    int bh = row / TT;
    int b  = bh / HH;
    float* wr = wbuf + (size_t)row * TT;
    const float* dr = dists + ((size_t)b * TT + i) * TT;
    int tid = threadIdx.x;
    __shared__ float red[256];
    float l[8];
    float mx = -3.0e38f;
#pragma unroll
    for (int r = 0; r < 8; ++r) {
        int j = r * 256 + tid;
        float v = (j <= i) ? wr[j] : -3.0e38f;
        l[r] = v; mx = fmaxf(mx, v);
    }
    red[tid] = mx; __syncthreads();
    for (int off = 128; off > 0; off >>= 1) {
        if (tid < off) red[tid] = fmaxf(red[tid], red[tid + off]);
        __syncthreads();
    }
    mx = red[0]; __syncthreads();
    float s = 0.f;
#pragma unroll
    for (int r = 0; r < 8; ++r) {
        int j = r * 256 + tid;
        float e = (j <= i) ? __expf(l[r] - mx) : 0.f;
        l[r] = e; s += e;
    }
    red[tid] = s; __syncthreads();
    for (int off = 128; off > 0; off >>= 1) {
        if (tid < off) red[tid] += red[tid + off];
        __syncthreads();
    }
    float inv = 1.0f / red[0]; __syncthreads();
    float ed = 0.f, ef = 0.f;
#pragma unroll
    for (int r = 0; r < 8; ++r) {
        int j = r * 256 + tid;
        float wv = l[r] * inv;
        wr[j] = wv;
        if (j <= i) {
            ed += wv * dr[j];
            ef -= wv * __logf(wv + 1e-9f);
        }
    }
    red[tid] = ed; __syncthreads();
    for (int off = 128; off > 0; off >>= 1) {
        if (tid < off) red[tid] += red[tid + off];
        __syncthreads();
    }
    float edT = red[0]; __syncthreads();
    red[tid] = ef; __syncthreads();
    for (int off = 128; off > 0; off >>= 1) {
        if (tid < off) red[tid] += red[tid + off];
        __syncthreads();
    }
    if (tid == 0) {
        const float sc = 1.0f / (float)(BB * HH * TT);
        atomicAdd(&scal[0], edT * sc);
        atomicAdd(&scal[1], red[0] * sc);
    }
}

// ---------------- attn = w @ v, causally truncated K-loop, -> attn_bf[b][t][h*64+dh] -------
__global__ void k_attnv(const float* __restrict__ wbuf, const bf16* __restrict__ vT,
                        bf16* __restrict__ attnb) {
    int wave = blockIdx.x * (blockDim.x >> 5) + (threadIdx.x >> 5);
    int lane = threadIdx.x & 31;
    const int tilesPerBH = NT_T * (DHD / 16);   // 128*4 = 512
    int bh = wave / tilesPerBH;
    if (bh >= BB * HH) return;
    int t2 = wave % tilesPerBH;
    int it = t2 / 4, nt = t2 % 4;
    const float* Wr = wbuf + ((size_t)bh * TT + it * 16) * TT;
    const bf16* Bt  = vT + ((size_t)bh * DHD + nt * 16) * TT;
    int steps = (it + 2) / 2;                   // ceil((it+1)*16 / 32); tail rows are 0
    v8f acc = {};
    for (int kk = 0; kk < steps; ++kk) {
        v16bf a = ld_a_f32(Wr + kk * 32, TT, lane);
        v16bf bb = ld_b_bf16(Bt + kk * 32, TT, lane);
        acc = __builtin_amdgcn_wmma_f32_16x16x32_bf16(false, a, false, bb,
                                                      (short)0, acc, false, false);
    }
    int h = bh % HH, b = bh / HH;
    int mloc = C_MLOC(lane), nloc = C_NLOC(lane);
    int col = h * DHD + nt * 16 + nloc;
#pragma unroll
    for (int i = 0; i < 8; ++i) {
        int tg = it * 16 + mloc + i;
        attnb[((size_t)b * TT + tg) * DD + col] = (bf16)acc[i];
    }
}

__global__ void k_zero2(float* s) { if (threadIdx.x < 2 && blockIdx.x == 0) s[threadIdx.x] = 0.f; }

// ==============================================================================
extern "C" void kernel_launch(void* const* d_in, const int* in_sizes, int n_in,
                              void* d_out, int out_size, void* d_ws, size_t ws_size,
                              hipStream_t stream) {
    const float* x      = (const float*)d_in[0];
    const float* dists  = (const float*)d_in[1];
    const float* ln1_g  = (const float*)d_in[2];
    const float* ln1_b  = (const float*)d_in[3];
    const float* w_qkv  = (const float*)d_in[4];
    const float* w_proj = (const float*)d_in[5];
    const float* ln2_g  = (const float*)d_in[6];
    const float* ln2_b  = (const float*)d_in[7];
    const float* w1     = (const float*)d_in[8];
    const float* b1     = (const float*)d_in[9];
    const float* w2     = (const float*)d_in[10];
    const float* b2     = (const float*)d_in[11];

    float* out = (float*)d_out;
    // output layout: x | dist_e | flop_e | w | token_dists
    const size_t SC_OFF = (size_t)BB * TT * DD;                 // 4,194,304
    const size_t W_OFF  = SC_OFF + 2;
    const size_t TD_OFF = W_OFF + (size_t)BB * HH * TT * TT;
    float* xout  = out;
    float* scal  = out + SC_OFF;
    float* wbuf  = out + W_OFF;
    float* tdout = out + TD_OFF;

    // workspace carve-up (bf16, 256B aligned)
    char* p = (char*)d_ws;
    auto carve = [&](size_t elems) -> bf16* {
        bf16* r = (bf16*)p;
        p += ((elems * sizeof(bf16) + 255) / 256) * 256;
        return r;
    };
    bf16* wqkvT  = carve((size_t)D3 * DD);
    bf16* wprojT = carve((size_t)DD * DD);
    bf16* w1T    = carve((size_t)DF * DD);
    bf16* w2T    = carve((size_t)DD * DF);
    bf16* h1     = carve((size_t)BB * TT * DD);
    bf16* qh     = carve((size_t)BB * HH * TT * DHD);
    bf16* kh     = carve((size_t)BB * HH * TT * DHD);
    bf16* vh     = carve((size_t)BB * HH * TT * DHD);
    bf16* vT     = carve((size_t)BB * HH * TT * DHD);
    bf16* attnb  = carve((size_t)BB * TT * DD);
    bf16* h2     = carve((size_t)BB * TT * DD);
    bf16* f1     = carve((size_t)BB * TT * DF);

    const dim3 blk(256);

    // pass-through output + scalar init
    hipMemcpyAsync(tdout, dists, (size_t)BB * TT * TT * sizeof(float),
                   hipMemcpyDeviceToDevice, stream);
    k_zero2<<<1, blk, 0, stream>>>(scal);

    // weight precompute (convert + transpose to bf16 B^T)
    k_prep_wT<<<((size_t)DD * D3 + 255) / 256, blk, 0, stream>>>(w_qkv,  wqkvT,  DD, D3);
    k_prep_wT<<<((size_t)DD * DD + 255) / 256, blk, 0, stream>>>(w_proj, wprojT, DD, DD);
    k_prep_wT<<<((size_t)DD * DF + 255) / 256, blk, 0, stream>>>(w1,     w1T,    DD, DF);
    k_prep_wT<<<((size_t)DF * DD + 255) / 256, blk, 0, stream>>>(w2,     w2T,    DF, DD);

    // LN1 -> h1 (bf16)
    k_layernorm<<<BB * TT, blk, 0, stream>>>(x, ln1_g, ln1_b, h1);

    // QKV GEMM: async-LDS + TDM staged, scatter into per-head q/k/v and vT
    k_gemm_staged<3><<<dim3(MT_ALL / 4, D3 / 64), blk, 0, stream>>>(
        h1, wqkvT, DD, D3, nullptr, nullptr, nullptr, nullptr, qh, kh, vh, vT);

    // logits = QK^T/sqrt(dh) - dist (lower triangle tiles only)
    {
        int waves = BB * HH * TRI;
        k_logits<<<(waves + 7) / 8, blk, 0, stream>>>(qh, kh, dists, wbuf);
    }
    // softmax rows + energies
    k_softmax<<<BB * HH * TT, blk, 0, stream>>>(wbuf, dists, scal);
    // attn = w @ v
    {
        int waves = BB * HH * NT_T * (DHD / 16);
        k_attnv<<<(waves + 7) / 8, blk, 0, stream>>>(wbuf, vT, attnb);
    }
    // proj + residual (staged)
    k_gemm_staged<0><<<dim3(MT_ALL / 4, DD / 64), blk, 0, stream>>>(
        attnb, wprojT, DD, DD, x, nullptr, xout, nullptr,
        nullptr, nullptr, nullptr, nullptr);
    // LN2 -> h2
    k_layernorm<<<BB * TT, blk, 0, stream>>>(xout, ln2_g, ln2_b, h2);
    // FFN1 (staged, GELU fused)
    k_gemm_staged<1><<<dim3(MT_ALL / 4, DF / 64), blk, 0, stream>>>(
        h2, w1T, DD, DF, nullptr, b1, nullptr, f1,
        nullptr, nullptr, nullptr, nullptr);
    // FFN2 (staged, bias + residual)
    k_gemm_staged<2><<<dim3(MT_ALL / 4, DD / 64), blk, 0, stream>>>(
        f1, w2T, DF, DD, nullptr, b2, xout, nullptr,
        nullptr, nullptr, nullptr, nullptr);
}
